// DynamicTransformerBlock_65249143160967
// MI455X (gfx1250) — matrix operand order repaired and checked by reading, not compiled, not run on hardware
//
#include <hip/hip_runtime.h>
#include <hip/hip_bf16.h>
#include <math.h>

#define E      1024
#define NHEAD  16
#define HD     64
#define HID    4096
#define BATCH  8
#define SEQ    1024
#define M_TOK  (BATCH * SEQ)   // 8192 tokens

typedef __attribute__((ext_vector_type(16))) __bf16 bf16x16;
typedef __attribute__((ext_vector_type(8)))  float  f32x8;
typedef __attribute__((ext_vector_type(4)))  unsigned int u32x4;
typedef __attribute__((ext_vector_type(8)))  int    i32x8;
typedef __attribute__((ext_vector_type(4)))  int    i32x4;

union FragAB {
    bf16x16 v;
    uint4   q[2];
};

#if defined(__HIP_DEVICE_COMPILE__) && __has_builtin(__builtin_amdgcn_global_load_async_to_lds_b128)
#define HAVE_ASYNC_BUILTIN 1
#else
#define HAVE_ASYNC_BUILTIN 0
#endif

#if defined(__HIP_DEVICE_COMPILE__) && __has_builtin(__builtin_amdgcn_tensor_load_to_lds)
#define HAVE_TDM 1
#else
#define HAVE_TDM 0
#endif

__device__ __forceinline__ unsigned short f2bf(float f) {
    unsigned u = __builtin_bit_cast(unsigned, f);
    u += 0x7FFFu + ((u >> 16) & 1u);          // round-to-nearest-even
    return (unsigned short)(u >> 16);
}
__device__ __forceinline__ float bf2f(unsigned short h) {
    unsigned u = ((unsigned)h) << 16;
    return __builtin_bit_cast(float, u);
}

__device__ __forceinline__ f32x8 wmma_bf16(bf16x16 a, bf16x16 b, f32x8 c) {
    return __builtin_amdgcn_wmma_f32_16x16x32_bf16(
        false, a, false, b, (short)0, c, false, false);
}

__device__ __forceinline__ f32x8 zero8() {
    f32x8 z = {0.f, 0.f, 0.f, 0.f, 0.f, 0.f, 0.f, 0.f};
    return z;
}

// 16-byte async global -> LDS copy (per lane), tracked by ASYNCcnt
__device__ __forceinline__ void async_ld16(void* lds, const void* g) {
#if HAVE_ASYNC_BUILTIN
    typedef __attribute__((address_space(1))) i32x4 gvec_t;   // global
    typedef __attribute__((address_space(3))) i32x4 lvec_t;   // LDS
    __builtin_amdgcn_global_load_async_to_lds_b128(
        (gvec_t*)(unsigned long long)(size_t)g,
        (lvec_t*)(unsigned)(size_t)lds,
        0, 0);
#else
    asm volatile("global_load_async_to_lds_b128 %0, %1, off"
                 :: "v"((unsigned)(unsigned long long)(size_t)lds),
                    "v"(g)
                 : "memory");
#endif
}

__device__ __forceinline__ void wait_async0() {
#if defined(__HIP_DEVICE_COMPILE__) && __has_builtin(__builtin_amdgcn_s_wait_asynccnt)
    __builtin_amdgcn_s_wait_asynccnt(0);
#else
    asm volatile("s_wait_asynccnt 0x0" ::: "memory");
#endif
}

__device__ __forceinline__ void wait_tensor0() {
#if defined(__HIP_DEVICE_COMPILE__) && __has_builtin(__builtin_amdgcn_s_wait_tensorcnt)
    __builtin_amdgcn_s_wait_tensorcnt(0);
#else
    asm volatile("s_wait_tensorcnt 0x0" ::: "memory");
#endif
}

#if HAVE_TDM
// TDM 2-D tile load: rows x cols (bf16), global row stride = stride_elems,
// packed row-major into LDS at `lds`. One instruction, tracked by TENSORcnt.
__device__ __forceinline__ void tdm_load_2d(void* lds, const void* gptr,
                                            unsigned rows, unsigned cols,
                                            unsigned stride_elems) {
    unsigned lds_addr = (unsigned)(unsigned long long)(size_t)lds;
    unsigned long long ga = (unsigned long long)(size_t)gptr;
    unsigned td0 = stride_elems;          // tensor dim0 (elements)
    unsigned td1 = 1u << 20;              // large: tile always in-bounds
    u32x4 g0;
    g0[0] = 1u;                                               // count=1
    g0[1] = lds_addr;                                         // lds_addr
    g0[2] = (unsigned)(ga & 0xFFFFFFFFu);                     // gaddr[31:0]
    g0[3] = (unsigned)((ga >> 32) & 0x01FFFFFFu) | 0x80000000u; // gaddr[56:32] | type=2
    i32x8 g1;
    g1[0] = 0x10000;                                          // data_size=1 (2B)
    g1[1] = (int)((td0 & 0xFFFFu) << 16);                     // tensor_dim0[15:0]
    g1[2] = (int)((td0 >> 16) | ((td1 & 0xFFFFu) << 16));     // dim0[31:16] | dim1[15:0]
    g1[3] = (int)((td1 >> 16) | (cols << 16));                // dim1[31:16] | tile_dim0
    g1[4] = (int)rows;                                        // tile_dim1 (tile_dim2=0)
    g1[5] = (int)stride_elems;                                // dim0_stride[31:0]
    g1[6] = 0;
    g1[7] = 0;
    i32x4 z4 = {0, 0, 0, 0};
    i32x8 z8 = {0, 0, 0, 0, 0, 0, 0, 0};
    __builtin_amdgcn_tensor_load_to_lds(g0, g1, z4, z4, z8, 0);
}
#endif

// ---------------------------------------------------------------- weights cvt
__global__ void cvt_f32_bf16(const float* __restrict__ in,
                             unsigned short* __restrict__ out, int n) {
    int i = blockIdx.x * blockDim.x + threadIdx.x;
    int stride = gridDim.x * blockDim.x;
    for (; i < n; i += stride) out[i] = f2bf(in[i]);
}

// ---------------------------------------------------------------- layernorm
__global__ __launch_bounds__(256) void layernorm_bf16(
    const float* __restrict__ x, const float* __restrict__ w,
    const float* __restrict__ b, unsigned short* __restrict__ out) {
    __shared__ float s1[256];
    __shared__ float s2[256];
    int row = blockIdx.x;
    int tid = threadIdx.x;
    int col = tid * 4;
    float4 xv = *(const float4*)(x + (size_t)row * E + col);
    float sum = xv.x + xv.y + xv.z + xv.w;
    float sq  = xv.x * xv.x + xv.y * xv.y + xv.z * xv.z + xv.w * xv.w;
    s1[tid] = sum; s2[tid] = sq;
    __syncthreads();
    for (int off = 128; off > 0; off >>= 1) {
        if (tid < off) { s1[tid] += s1[tid + off]; s2[tid] += s2[tid + off]; }
        __syncthreads();
    }
    float mu   = s1[0] * (1.0f / E);
    float var  = s2[0] * (1.0f / E) - mu * mu;
    float rstd = rsqrtf(var + 1e-5f);
    float vx[4] = {xv.x, xv.y, xv.z, xv.w};
#pragma unroll
    for (int j = 0; j < 4; ++j) {
        float v = (vx[j] - mu) * rstd * w[col + j] + b[col + j];
        out[(size_t)row * E + col + j] = f2bf(v);
    }
}

// ---------------------------------------------------------------- GEMM
// C[M,N] = act(A[M,K] @ W[N,K]^T + bias) (+residual). 128x64 tile, 128 thr.
// Double-buffered LDS; A staged by async-to-LDS, W staged by TDM (wave 0).
__global__ __launch_bounds__(128) void gemm_bf16(
    const unsigned short* __restrict__ A,   // [M,K] bf16
    const unsigned short* __restrict__ W,   // [N,K] bf16
    const float* __restrict__ bias,         // [N]
    const float* __restrict__ resid,        // [M,N] f32 or null
    float* __restrict__ Cf,                 // f32 out (or null)
    unsigned short* __restrict__ Cb,        // bf16 out (or null)
    int M, int N, int K, int applyGelu) {
    __shared__ unsigned short sA[2][128][32];   // 16 KB
    __shared__ unsigned short sW[2][64][32];    //  8 KB
    const int n0 = blockIdx.x * 64;
    const int m0 = blockIdx.y * 128;
    const int tid = threadIdx.x;
    const int l = tid & 31;
    const int w = tid >> 5;

    f32x8 acc[2][4];
#pragma unroll
    for (int mt = 0; mt < 2; ++mt)
#pragma unroll
        for (int nt = 0; nt < 4; ++nt) acc[mt][nt] = zero8();

    auto stageA = [&](int buf, int k0) {
        const unsigned short* ag = A + (size_t)(m0 + tid) * K + k0;
        unsigned short* lp = &sA[buf][tid][0];
        async_ld16(lp,      ag);
        async_ld16(lp + 8,  ag + 8);
        async_ld16(lp + 16, ag + 16);
        async_ld16(lp + 24, ag + 24);
    };
    auto stageW = [&](int buf, int k0) {
#if HAVE_TDM
        if (w == 0)
            tdm_load_2d(&sW[buf][0][0], W + (size_t)n0 * K + k0,
                        /*rows=*/64, /*cols=*/32, /*stride=*/(unsigned)K);
#else
        const unsigned short* wg =
            W + (size_t)(n0 + (tid >> 1)) * K + k0 + (tid & 1) * 16;
        unsigned short* lp = &sW[buf][tid >> 1][(tid & 1) * 16];
        async_ld16(lp,     wg);
        async_ld16(lp + 8, wg + 8);
#endif
    };

    const int nk = K / 32;
    stageA(0, 0);
    stageW(0, 0);
    for (int t = 0; t < nk; ++t) {
        wait_async0();
#if HAVE_TDM
        if (w == 0) wait_tensor0();
#endif
        __syncthreads();
        if (t + 1 < nk) { stageA((t + 1) & 1, (t + 1) * 32); stageW((t + 1) & 1, (t + 1) * 32); }
        const int buf = t & 1;
        // A fragments for this wave's two 16-row subtiles
        const int o1 = (l >= 16) ? 8 : 0;
        FragAB af[2];
#pragma unroll
        for (int mt = 0; mt < 2; ++mt) {
            int am = w * 32 + mt * 16 + (l & 15);
            af[mt].q[0] = *(const uint4*)&sA[buf][am][o1];
            af[mt].q[1] = *(const uint4*)&sA[buf][am][o1 + 16];
        }
        const int bo = (l >= 16) ? 8 : 0;
#pragma unroll
        for (int nt = 0; nt < 4; ++nt) {
            FragAB bfr;
            int bn = nt * 16 + (l & 15);
            bfr.q[0] = *(const uint4*)&sW[buf][bn][bo];
            bfr.q[1] = *(const uint4*)&sW[buf][bn][bo + 16];
            acc[0][nt] = wmma_bf16(af[0].v, bfr.v, acc[0][nt]);
            acc[1][nt] = wmma_bf16(af[1].v, bfr.v, acc[1][nt]);
        }
    }
    // epilogue
#pragma unroll
    for (int mt = 0; mt < 2; ++mt)
#pragma unroll
    for (int nt = 0; nt < 4; ++nt) {
        int n = n0 + nt * 16 + (l & 15);
        float bv = bias[n];
#pragma unroll
        for (int i = 0; i < 8; ++i) {
            int m = m0 + w * 32 + mt * 16 + i + ((l >= 16) ? 8 : 0);
            float v = acc[mt][nt][i] + bv;
            if (applyGelu) v = 0.5f * v * (1.0f + erff(v * 0.70710678118f));
            if (resid) v += resid[(size_t)m * N + n];
            if (Cf) Cf[(size_t)m * N + n] = v;
            else    Cb[(size_t)m * N + n] = f2bf(v);
        }
    }
}

// ---------------------------------------------------------------- attention
__global__ __launch_bounds__(128) void attention_kernel(
    const unsigned short* __restrict__ qkv,   // [M_TOK, 3E] bf16
    unsigned short* __restrict__ ctx) {       // [M_TOK, E]  bf16
    __shared__ unsigned short sQ[16][64];          // 2 KB
    __shared__ unsigned short sK[4][16][64];       // 8 KB (wave-private tiles)
    __shared__ unsigned short sS[16][1024];        // 32 KB score/prob strip
    __shared__ unsigned short sVt[64][32];         // 4 KB transposed V chunk
    __shared__ float sSum[16];

    int qt = blockIdx.x;
    int h  = blockIdx.y;
    int b  = blockIdx.z;
    int tid = threadIdx.x;
    int l = tid & 31;
    int w = tid >> 5;
    size_t baseTok = (size_t)b * SEQ;
    const size_t qoff = (size_t)h * HD;
    const size_t koff = E + (size_t)h * HD;
    const size_t voff = 2 * (size_t)E + (size_t)h * HD;

    // async-stage Q tile 16x64
    {
        int r = tid >> 3, c = (tid & 7) * 8;
        async_ld16(&sQ[r][c],
                   qkv + (baseTok + qt * 16 + r) * 3 * E + qoff + c);
    }
    wait_async0();
    __syncthreads();

    // ---- pass 1: scores ----
    for (int iter = 0; iter < 16; ++iter) {
        int kc = iter * 64 + w * 16;   // this wave's 16 keys
        int kr = l >> 1, kcol = (l & 1) * 32;
        const unsigned short* kp =
            qkv + (baseTok + kc + kr) * 3 * E + koff + kcol;
        unsigned short* lp = &sK[w][kr][kcol];
        async_ld16(lp,      kp);
        async_ld16(lp + 8,  kp + 8);
        async_ld16(lp + 16, kp + 16);
        async_ld16(lp + 24, kp + 24);
        wait_async0();                 // wave-private tile: own wait suffices
        f32x8 acc = zero8();
#pragma unroll
        for (int s = 0; s < 2; ++s) {
            FragAB af, bfr;
            int am = l & 15;
            int o1 = s * 32 + ((l >= 16) ? 8 : 0);
            af.q[0] = *(const uint4*)&sQ[am][o1];
            af.q[1] = *(const uint4*)&sQ[am][o1 + 16];
            int bn = l & 15;
            int bo = s * 32 + ((l >= 16) ? 8 : 0);
            bfr.q[0] = *(const uint4*)&sK[w][bn][bo];
            bfr.q[1] = *(const uint4*)&sK[w][bn][bo + 16];
            acc = wmma_bf16(af.v, bfr.v, acc);
        }
#pragma unroll
        for (int i = 0; i < 8; ++i) {
            int qr = i + ((l >= 16) ? 8 : 0);
            sS[qr][kc + (l & 15)] = f2bf(acc[i] * 0.125f);  // HD^-0.5
        }
    }
    __syncthreads();

    // ---- pass 2: softmax (wave w handles rows w*4..w*4+3) ----
    for (int rr = 0; rr < 4; ++rr) {
        int r = w * 4 + rr;
        float vals[32];
        float mx = -1e30f;
#pragma unroll
        for (int j = 0; j < 32; ++j) {
            float v = bf2f(sS[r][l + j * 32]);
            vals[j] = v;
            mx = fmaxf(mx, v);
        }
#pragma unroll
        for (int o = 16; o > 0; o >>= 1) mx = fmaxf(mx, __shfl_xor(mx, o, 32));
        float sum = 0.f;
#pragma unroll
        for (int j = 0; j < 32; ++j) {
            float e = __expf(vals[j] - mx);
            sum += e;
            sS[r][l + j * 32] = f2bf(e);
        }
#pragma unroll
        for (int o = 16; o > 0; o >>= 1) sum += __shfl_xor(sum, o, 32);
        if (l == 0) sSum[r] = sum;
    }
    __syncthreads();

    // ---- pass 3: P @ V ----
    f32x8 oacc = zero8();
    for (int kc = 0; kc < SEQ; kc += 32) {
#pragma unroll
        for (int rep = 0; rep < 2; ++rep) {
            int idx = tid + rep * 128;
            int k = idx >> 3;
            int dc = (idx & 7) * 8;
            uint4 vv = *(const uint4*)(qkv + (baseTok + kc + k) * 3 * E + voff + dc);
            const unsigned short* pv = (const unsigned short*)&vv;
#pragma unroll
            for (int j = 0; j < 8; ++j) sVt[dc + j][k] = pv[j];
        }
        __syncthreads();
        FragAB af, bfr;
        int am = l & 15;
        int o1 = kc + ((l >= 16) ? 8 : 0);
        af.q[0] = *(const uint4*)&sS[am][o1];
        af.q[1] = *(const uint4*)&sS[am][o1 + 16];
        int d = w * 16 + (l & 15);
        int bo = (l >= 16) ? 8 : 0;
        bfr.q[0] = *(const uint4*)&sVt[d][bo];
        bfr.q[1] = *(const uint4*)&sVt[d][bo + 16];
        oacc = wmma_bf16(af.v, bfr.v, oacc);
        __syncthreads();
    }
#pragma unroll
    for (int i = 0; i < 8; ++i) {
        int qr = i + ((l >= 16) ? 8 : 0);
        int d = w * 16 + (l & 15);
        float v = oacc[i] / sSum[qr];
        ctx[(baseTok + qt * 16 + qr) * E + (size_t)h * HD + d] = f2bf(v);
    }
}

// ---------------------------------------------------------------- launch
extern "C" void kernel_launch(void* const* d_in, const int* in_sizes, int n_in,
                              void* d_out, int out_size, void* d_ws, size_t ws_size,
                              hipStream_t stream) {
    (void)in_sizes; (void)n_in; (void)out_size; (void)ws_size;
    const float* x     = (const float*)d_in[0];
    const float* ln1w  = (const float*)d_in[1];
    const float* ln1b  = (const float*)d_in[2];
    const float* wqkv  = (const float*)d_in[3];
    const float* bqkv  = (const float*)d_in[4];
    const float* wproj = (const float*)d_in[5];
    const float* bproj = (const float*)d_in[6];
    const float* ln2w  = (const float*)d_in[7];
    const float* ln2b  = (const float*)d_in[8];
    const float* wfc1  = (const float*)d_in[9];
    const float* bfc1  = (const float*)d_in[10];
    const float* wfc2  = (const float*)d_in[11];
    const float* bfc2  = (const float*)d_in[12];
    float* out = (float*)d_out;

    char* ws = (char*)d_ws;
    size_t off = 0;
    auto alloc = [&](size_t bytes) -> void* {
        void* p = ws + off;
        off += (bytes + 255) & ~(size_t)255;
        return p;
    };
    unsigned short* hbf    = (unsigned short*)alloc((size_t)M_TOK * E * 2);
    unsigned short* wqkvb  = (unsigned short*)alloc((size_t)3 * E * E * 2);
    unsigned short* wprojb = (unsigned short*)alloc((size_t)E * E * 2);
    unsigned short* wfc1b  = (unsigned short*)alloc((size_t)HID * E * 2);
    unsigned short* wfc2b  = (unsigned short*)alloc((size_t)E * HID * 2);
    unsigned short* bigbuf = (unsigned short*)alloc((size_t)M_TOK * HID * 2);
    unsigned short* ctx    = (unsigned short*)alloc((size_t)M_TOK * E * 2);
    float*          x1     = (float*)alloc((size_t)M_TOK * E * 4);

    cvt_f32_bf16<<<512, 256, 0, stream>>>(wqkv,  wqkvb,  3 * E * E);
    cvt_f32_bf16<<<512, 256, 0, stream>>>(wproj, wprojb, E * E);
    cvt_f32_bf16<<<512, 256, 0, stream>>>(wfc1,  wfc1b,  HID * E);
    cvt_f32_bf16<<<512, 256, 0, stream>>>(wfc2,  wfc2b,  E * HID);

    layernorm_bf16<<<M_TOK, 256, 0, stream>>>(x, ln1w, ln1b, hbf);

    gemm_bf16<<<dim3(3 * E / 64, M_TOK / 128), 128, 0, stream>>>(
        hbf, wqkvb, bqkv, nullptr, nullptr, bigbuf, M_TOK, 3 * E, E, 0);

    attention_kernel<<<dim3(SEQ / 16, NHEAD, BATCH), 128, 0, stream>>>(bigbuf, ctx);

    gemm_bf16<<<dim3(E / 64, M_TOK / 128), 128, 0, stream>>>(
        ctx, wprojb, bproj, x, x1, nullptr, M_TOK, E, E, 0);

    layernorm_bf16<<<M_TOK, 256, 0, stream>>>(x1, ln2w, ln2b, hbf);

    gemm_bf16<<<dim3(HID / 64, M_TOK / 128), 128, 0, stream>>>(
        hbf, wfc1b, bfc1, nullptr, nullptr, bigbuf, M_TOK, HID, E, 1);

    gemm_bf16<<<dim3(E / 64, M_TOK / 128), 128, 0, stream>>>(
        bigbuf, wfc2b, bfc2, x1, out, nullptr, M_TOK, E, HID, 0);
}
